// DSAMultiHeadAttention_75316546503193
// MI455X (gfx1250) — matrix operand order repaired and checked by reading, not compiled, not run on hardware
//
#include <hip/hip_runtime.h>
#include <hip/hip_bf16.h>

// ---------------------------------------------------------------------------
// Sizes
// ---------------------------------------------------------------------------
#define BATCH 2
#define SEQ   4096
#define EMB   512
#define HEADS 8
#define HDIM  64
#define MROWS (BATCH * SEQ)   // 8192
#define LN_EPS 1e-5f

typedef __attribute__((ext_vector_type(8)))  __bf16 bf16x8;
typedef __attribute__((ext_vector_type(16))) __bf16 bf16x16;
typedef __attribute__((ext_vector_type(8)))  float  f32x8;

static __device__ __forceinline__ __bf16 f2bf(float f) {
    unsigned u = __builtin_bit_cast(unsigned, f);
    unsigned r = (u + 0x7FFFu + ((u >> 16) & 1u)) >> 16;
    unsigned short h = (unsigned short)r;
    return __builtin_bit_cast(__bf16, h);
}

// Load a 16-element fragment as two 16-byte chunks (p0 -> [0..7], p1 -> [8..15])
static __device__ __forceinline__ bf16x16 ld16(const __bf16* p0, const __bf16* p1) {
    bf16x8 a = *(const bf16x8*)p0;
    bf16x8 b = *(const bf16x8*)p1;
    bf16x16 r;
#pragma unroll
    for (int i = 0; i < 8; ++i) { r[i] = a[i]; r[i + 8] = b[i]; }
    return r;
}

static __device__ __forceinline__ f32x8 wmma_bf16(bf16x16 a, bf16x16 b, f32x8 c) {
    return __builtin_amdgcn_wmma_f32_16x16x32_bf16(false, a, false, b, (short)0, c,
                                                   false, false);
}

// A-fragment (16x32 bf16): row = lane&15, chunks at k0+8h and k0+16+8h
static __device__ __forceinline__ bf16x16 ldA(const __bf16* base, int row, int stride,
                                              int k0, int nn, int hh) {
    const __bf16* p = base + (size_t)(row + nn) * stride + k0 + 8 * hh;
    return ld16(p, p + 16);
}
// B-fragment (32x16 bf16) from column-major storage (bt[n][k] row-major):
// col = lane&15, chunks at k0+8h and k0+16+8h
static __device__ __forceinline__ bf16x16 ldB(const __bf16* bt, int col0, int stride,
                                              int k0, int nn, int hh) {
    const __bf16* p = bt + (size_t)(col0 + nn) * stride + k0 + 8 * hh;
    return ld16(p, p + 16);
}

// ---------------------------------------------------------------------------
// Stage 0: conversions
// ---------------------------------------------------------------------------
__global__ void cvt_bf16_kernel(const float* __restrict__ src, __bf16* __restrict__ dst,
                                int n) {
    int i = blockIdx.x * blockDim.x + threadIdx.x;
    if (i < n) dst[i] = f2bf(src[i]);
}

// w[in][out] (512x512) -> wt[out][in] bf16
__global__ void transpose_w_kernel(const float* __restrict__ w, __bf16* __restrict__ wt) {
    int idx = blockIdx.x * blockDim.x + threadIdx.x;   // 512*512 threads
    int o = idx >> 9, i = idx & 511;
    wt[(size_t)o * EMB + i] = f2bf(w[(size_t)i * EMB + o]);
}

// ---------------------------------------------------------------------------
// Stage 1: QKV projection.  grid (M/64, E/64, 3), block 128 (4 waves)
// ---------------------------------------------------------------------------
__global__ __launch_bounds__(128)
void qkv_gemm_kernel(const __bf16* __restrict__ xb, const __bf16* __restrict__ wT,
                     const float* __restrict__ bq, const float* __restrict__ bk,
                     const float* __restrict__ bv,
                     __bf16* __restrict__ qb, __bf16* __restrict__ kb,
                     __bf16* __restrict__ vtb) {
    const int z    = blockIdx.z;
    const __bf16* wt   = wT + (size_t)z * EMB * EMB;
    const float*  bias = (z == 0) ? bq : (z == 1) ? bk : bv;

    const int w    = threadIdx.x >> 5;
    const int lane = threadIdx.x & 31;
    const int hh   = lane >> 4;
    const int nn   = lane & 15;
    const int row0 = blockIdx.x * 64 + w * 16;
    const int col0 = blockIdx.y * 64;

    f32x8 acc[4] = {};
#pragma unroll 4
    for (int k0 = 0; k0 < EMB; k0 += 32) {
        bf16x16 a = ldA(xb, row0, EMB, k0, nn, hh);
#pragma unroll
        for (int jt = 0; jt < 4; ++jt) {
            bf16x16 b = ldB(wt, col0 + jt * 16, EMB, k0, nn, hh);
            acc[jt] = wmma_bf16(a, b, acc[jt]);
        }
    }
#pragma unroll
    for (int jt = 0; jt < 4; ++jt) {
        int c    = col0 + jt * 16 + nn;
        int head = c >> 6, d = c & 63;
#pragma unroll
        for (int r = 0; r < 8; ++r) {
            int m = row0 + r + 8 * hh;
            int bI = m >> 12, s = m & (SEQ - 1);
            __bf16 v = f2bf(acc[jt][r] + bias[c]);
            size_t bh = (size_t)bI * HEADS + head;
            if (z == 0)      qb [(bh * SEQ + s) * HDIM + d] = v;
            else if (z == 1) kb [(bh * SEQ + s) * HDIM + d] = v;
            else             vtb[(bh * HDIM + d) * SEQ + s] = v;
        }
    }
}

// ---------------------------------------------------------------------------
// Stage 2: flash attention.  grid (S/64, B*H), block 128 (4 waves, 16 rows each)
// ---------------------------------------------------------------------------
__global__ __launch_bounds__(128)
void flash_attn_kernel(const __bf16* __restrict__ qb, const __bf16* __restrict__ kb,
                       const __bf16* __restrict__ vtb, __bf16* __restrict__ ctx) {
    const int bh   = blockIdx.y;
    const int bI   = bh >> 3, head = bh & 7;
    const __bf16* Q  = qb  + (size_t)bh * SEQ * HDIM;
    const __bf16* K  = kb  + (size_t)bh * SEQ * HDIM;
    const __bf16* VT = vtb + (size_t)bh * HDIM * SEQ;

    const int w    = threadIdx.x >> 5;
    const int lane = threadIdx.x & 31;
    const int hh   = lane >> 4;
    const int nn   = lane & 15;
    const int qrow0 = blockIdx.x * 64 + w * 16;

    __shared__ __align__(16) __bf16 pst[4][16 * 72];
    __bf16* myp = &pst[w][0];

    bf16x16 qf[2];
#pragma unroll
    for (int t = 0; t < 2; ++t) qf[t] = ldA(Q, qrow0, HDIM, 32 * t, nn, hh);

    float mstate[8], lstate[8];
#pragma unroll
    for (int r = 0; r < 8; ++r) { mstate[r] = -1e30f; lstate[r] = 0.f; }
    f32x8 of[4] = {};

    for (int it = 0; it < SEQ / 64; ++it) {
        const int kb0 = it * 64;

        // S = Q * K^T  (16x64 tile, 8 WMMAs)
        f32x8 sf[4] = {};
#pragma unroll
        for (int t = 0; t < 2; ++t) {
            bf16x16 a = qf[t];
#pragma unroll
            for (int jt = 0; jt < 4; ++jt) {
                bf16x16 b = ldB(K, kb0 + jt * 16, HDIM, 32 * t, nn, hh);
                sf[jt] = wmma_bf16(a, b, sf[jt]);
            }
        }

        if (it + 1 < SEQ / 64) {   // pull next K/V tiles toward L0/L2
            __builtin_prefetch(K + (size_t)(kb0 + 64 + lane) * HDIM, 0, 1);
            __builtin_prefetch(K + (size_t)(kb0 + 96 + lane) * HDIM, 0, 1);
            __builtin_prefetch(VT + (size_t)(lane * 2) * SEQ + kb0 + 64, 0, 1);
            __builtin_prefetch(VT + (size_t)(lane * 2 + 1) * SEQ + kb0 + 64, 0, 1);
        }

        // online softmax per row (rows r+8h live across 16-lane halves)
#pragma unroll
        for (int r = 0; r < 8; ++r) {
            float mx = -1e30f;
#pragma unroll
            for (int jt = 0; jt < 4; ++jt) {
                float s = sf[jt][r] * 0.125f;   // 1/sqrt(64)
                sf[jt][r] = s;
                mx = fmaxf(mx, s);
            }
            mx = fmaxf(mx, __shfl_xor(mx, 1, 32));
            mx = fmaxf(mx, __shfl_xor(mx, 2, 32));
            mx = fmaxf(mx, __shfl_xor(mx, 4, 32));
            mx = fmaxf(mx, __shfl_xor(mx, 8, 32));
            float mnew = fmaxf(mstate[r], mx);
            float corr = __expf(mstate[r] - mnew);
            mstate[r] = mnew;
            float rs = 0.f;
#pragma unroll
            for (int jt = 0; jt < 4; ++jt) {
                float p = __expf(sf[jt][r] - mnew);
                sf[jt][r] = p;
                rs += p;
            }
            rs += __shfl_xor(rs, 1, 32);
            rs += __shfl_xor(rs, 2, 32);
            rs += __shfl_xor(rs, 4, 32);
            rs += __shfl_xor(rs, 8, 32);
            lstate[r] = lstate[r] * corr + rs;
#pragma unroll
            for (int dt = 0; dt < 4; ++dt) of[dt][r] = of[dt][r] * corr;
            // stage P (C-layout -> LDS row-major)
#pragma unroll
            for (int jt = 0; jt < 4; ++jt)
                myp[(r + 8 * hh) * 72 + jt * 16 + nn] = f2bf(sf[jt][r]);
        }

        // O += P * V  (P from LDS as A-fragments, V^T columns contiguous)
#pragma unroll
        for (int t = 0; t < 2; ++t) {
            const __bf16* pa = myp + nn * 72 + 32 * t + 8 * hh;
            bf16x16 a = ld16(pa, pa + 16);
#pragma unroll
            for (int dt = 0; dt < 4; ++dt) {
                bf16x16 b = ldB(VT, dt * 16, SEQ, kb0 + 32 * t, nn, hh);
                of[dt] = wmma_bf16(a, b, of[dt]);
            }
        }
    }

    // normalize + write ctx[m][head*64+d] (bf16, row-major M x 512)
#pragma unroll
    for (int dt = 0; dt < 4; ++dt) {
        int d = dt * 16 + nn;
#pragma unroll
        for (int r = 0; r < 8; ++r) {
            int srow = qrow0 + r + 8 * hh;
            float o = of[dt][r] / lstate[r];
            size_t idx = ((size_t)bI * SEQ + srow) * EMB + head * HDIM + d;
            ctx[idx] = f2bf(o);
        }
    }
}

// ---------------------------------------------------------------------------
// Stage 3: output projection + bias + residual.  grid (M/64, E/64), block 128
// ---------------------------------------------------------------------------
__global__ __launch_bounds__(128)
void out_gemm_kernel(const __bf16* __restrict__ ctxb, const __bf16* __restrict__ woT,
                     const float* __restrict__ bo, const float* __restrict__ xres,
                     float* __restrict__ y) {
    const int w    = threadIdx.x >> 5;
    const int lane = threadIdx.x & 31;
    const int hh   = lane >> 4;
    const int nn   = lane & 15;
    const int row0 = blockIdx.x * 64 + w * 16;
    const int col0 = blockIdx.y * 64;

    f32x8 acc[4] = {};
#pragma unroll 4
    for (int k0 = 0; k0 < EMB; k0 += 32) {
        bf16x16 a = ldA(ctxb, row0, EMB, k0, nn, hh);
#pragma unroll
        for (int jt = 0; jt < 4; ++jt) {
            bf16x16 b = ldB(woT, col0 + jt * 16, EMB, k0, nn, hh);
            acc[jt] = wmma_bf16(a, b, acc[jt]);
        }
    }
#pragma unroll
    for (int jt = 0; jt < 4; ++jt) {
        int c = col0 + jt * 16 + nn;
        float bc = bo[c];
#pragma unroll
        for (int r = 0; r < 8; ++r) {
            int m = row0 + r + 8 * hh;
            size_t idx = (size_t)m * EMB + c;
            y[idx] = acc[jt][r] + bc + xres[idx];
        }
    }
}

// ---------------------------------------------------------------------------
// Stage 4: LayerNorm, one wave per row.  grid M/8, block 256 (8 waves)
// ---------------------------------------------------------------------------
__global__ __launch_bounds__(256)
void ln_kernel(const float* __restrict__ y, const float* __restrict__ g,
               const float* __restrict__ b, float* __restrict__ out) {
    const int w    = threadIdx.x >> 5;
    const int lane = threadIdx.x & 31;
    const int row  = blockIdx.x * 8 + w;
    const float* yr = y + (size_t)row * EMB;

    float v[16];
    float s = 0.f, s2 = 0.f;
#pragma unroll
    for (int j = 0; j < 4; ++j) {
        float4 q = *(const float4*)(yr + lane * 16 + j * 4);
        v[j * 4 + 0] = q.x; v[j * 4 + 1] = q.y; v[j * 4 + 2] = q.z; v[j * 4 + 3] = q.w;
        s  += q.x + q.y + q.z + q.w;
        s2 += q.x * q.x + q.y * q.y + q.z * q.z + q.w * q.w;
    }
#pragma unroll
    for (int m = 16; m >= 1; m >>= 1) {
        s  += __shfl_xor(s,  m, 32);
        s2 += __shfl_xor(s2, m, 32);
    }
    float mu  = s * (1.f / EMB);
    float var = s2 * (1.f / EMB) - mu * mu;
    float inv = rsqrtf(var + LN_EPS);
    float* orow = out + (size_t)row * EMB;
#pragma unroll
    for (int i = 0; i < 16; ++i) {
        int c = lane * 16 + i;
        orow[c] = (v[i] - mu) * inv * g[c] + b[c];
    }
}

// ---------------------------------------------------------------------------
// Host launcher
// ---------------------------------------------------------------------------
extern "C" void kernel_launch(void* const* d_in, const int* in_sizes, int n_in,
                              void* d_out, int out_size, void* d_ws, size_t ws_size,
                              hipStream_t stream) {
    const float* x    = (const float*)d_in[0];
    const float* wq   = (const float*)d_in[1];
    const float* bq   = (const float*)d_in[2];
    const float* wk   = (const float*)d_in[3];
    const float* bk   = (const float*)d_in[4];
    const float* wv   = (const float*)d_in[5];
    const float* bv   = (const float*)d_in[6];
    const float* wo   = (const float*)d_in[7];
    const float* bo   = (const float*)d_in[8];
    const float* ln_g = (const float*)d_in[9];
    const float* ln_b = (const float*)d_in[10];
    float* out = (float*)d_out;

    char* ws = (char*)d_ws;
    const size_t SZ_XB  = (size_t)MROWS * EMB * 2;            // 8 MB
    const size_t SZ_W   = (size_t)EMB * EMB * 2;              // 512 KB each
    const size_t SZ_QKV = (size_t)BATCH * HEADS * SEQ * HDIM * 2;  // 8 MB each
    const size_t SZ_CTX = (size_t)MROWS * EMB * 2;            // 8 MB
    __bf16* xb  = (__bf16*)(ws);
    __bf16* wT  = (__bf16*)(ws + SZ_XB);                      // 4 matrices
    __bf16* qb  = (__bf16*)(ws + SZ_XB + 4 * SZ_W);
    __bf16* kb  = (__bf16*)(ws + SZ_XB + 4 * SZ_W + SZ_QKV);
    __bf16* vtb = (__bf16*)(ws + SZ_XB + 4 * SZ_W + 2 * SZ_QKV);
    __bf16* ctx = (__bf16*)(ws + SZ_XB + 4 * SZ_W + 3 * SZ_QKV);
    float*  yws = (float*) (ws + SZ_XB + 4 * SZ_W + 3 * SZ_QKV + SZ_CTX);

    // stage 0: conversions
    int nx = MROWS * EMB;
    cvt_bf16_kernel<<<(nx + 255) / 256, 256, 0, stream>>>(x, xb, nx);
    transpose_w_kernel<<<(EMB * EMB) / 256, 256, 0, stream>>>(wq, wT + 0 * EMB * EMB);
    transpose_w_kernel<<<(EMB * EMB) / 256, 256, 0, stream>>>(wk, wT + 1 * EMB * EMB);
    transpose_w_kernel<<<(EMB * EMB) / 256, 256, 0, stream>>>(wv, wT + 2 * EMB * EMB);
    transpose_w_kernel<<<(EMB * EMB) / 256, 256, 0, stream>>>(wo, wT + 3 * EMB * EMB);

    // stage 1: QKV projections
    qkv_gemm_kernel<<<dim3(MROWS / 64, EMB / 64, 3), 128, 0, stream>>>(
        xb, wT, bq, bk, bv, qb, kb, vtb);

    // stage 2: flash attention
    flash_attn_kernel<<<dim3(SEQ / 64, BATCH * HEADS), 128, 0, stream>>>(
        qb, kb, vtb, ctx);

    // stage 3: output projection + residual
    out_gemm_kernel<<<dim3(MROWS / 64, EMB / 64), 128, 0, stream>>>(
        ctx, wT + 3 * EMB * EMB, bo, x, yws);

    // stage 4: LayerNorm
    ln_kernel<<<MROWS / 8, 256, 0, stream>>>(yws, ln_g, ln_b, out);
}
